// GridGNNLayer_63694365000285
// MI455X (gfx1250) — matrix operand order repaired and checked by reading, not compiled, not run on hardware
//
#include <hip/hip_runtime.h>

// ---------------------------------------------------------------------------
// GridGNN layer for MI455X (gfx1250, wave32, WMMA + TDM).
// Node-network convs (95% of FLOPs) run as im2col GEMMs on
// v_wmma_f32_16x16x32_f16 with f32 accumulation; packed weights are staged
// into LDS by the Tensor Data Mover (tensor_load_to_lds + s_wait_tensorcnt).
// Small memory-bound stages run on VALU f32.
// ---------------------------------------------------------------------------

typedef _Float16 half16 __attribute__((ext_vector_type(16)));
typedef float    float8 __attribute__((ext_vector_type(8)));
typedef unsigned int u32x4 __attribute__((ext_vector_type(4)));
typedef int          i32x8 __attribute__((ext_vector_type(8)));
typedef int          i32x4 __attribute__((ext_vector_type(4)));

union Frag { half16 h; uint4 q[2]; };

// problem constants (match reference setup_inputs)
constexpr int N_IMG = 256, H = 64, W = 64;
constexpr int E_EDGE = 4096, EE = 4, HID = 64, EMB = 32;
constexpr int HS = 32, WSM = 32;          // downsampled h,w
// node-net GEMM geometry
constexpr int CP1 = 40;                   // 36 input channels padded to 40 (8-aligned)
constexpr int K1R = 9 * CP1;              // 360 real K for conv1
constexpr int K1P = 384;                  // padded to 12 * 32
constexpr int CP2 = 64;                   // conv2 input channels (exact)
constexpr int K2  = 9 * CP2;              // 576 = 18 * 32
constexpr int TIL = 16;                   // output tile edge
constexpr int IR  = 20;                   // staged input region (TIL + 4)
constexpr int HR  = 18;                   // hidden region (TIL + 2)

// node-kernel dynamic LDS layout (no static LDS in this kernel -> dynamic
// segment starts at LDS offset 0, so these offsets ARE the hardware LDS
// addresses used in the TDM descriptor)
constexpr int S_IN_BYTES  = IR * IR * CP1 * 2;   // 32000
constexpr int S_W_BYTES   = HID * K1P * 2;       // 49152 (>= 32*576*2)
constexpr int S_HID_BYTES = 336 * CP2 * 2;       // 43008 (21 M-tiles of 16 rows)
constexpr int NODE_LDS    = S_IN_BYTES + S_W_BYTES + S_HID_BYTES; // 124160 B -> 2 WG/WGP

// workspace offsets (bytes, 16B aligned)
constexpr size_t XD_OFF  = 0;
constexpr size_t E2_OFF  = XD_OFF  + (size_t)N_IMG * EE * HS * WSM * 4;   //  +4 MB
constexpr size_t AGG_OFF = E2_OFF  + (size_t)E_EDGE * EE * HS * WSM * 4;  //  +64 MB
constexpr size_t UP_OFF  = AGG_OFF + (size_t)N_IMG * EE * HS * WSM * 4;   //  +4 MB
constexpr size_t W1_OFF  = UP_OFF  + (size_t)N_IMG * EE * H * W * 4;      //  +16 MB
constexpr size_t W2_OFF  = W1_OFF  + (size_t)HID * K1P * 2;               //  +48 KB
// total ~88.1 MB of d_ws

// ---------------------------------------------------------------------------
// TDM helper: 1-D DMA copy global -> LDS (n8 x 8-byte elements).
// D# encoding per CDNA5 ISA:
//   group0: count=1 (valid), lds_addr, global_addr[56:0], type=2
//   group1: data_size=3 (8B), tensor_dim0=n8, tile_dim0=n8, tile_dim1=0 (1-D)
//   groups 2/3 (and trailing group word): zero (no higher dims, no iteration)
// Issue from ONE wave only; completion via s_wait_tensorcnt 0.
// NOTE: this toolchain exposes the 6-arg builtin:
//   (uint32x4, int32x8, int32x4, int32x4, int32x8, i32 cpol)
// ---------------------------------------------------------------------------
static __device__ __forceinline__
void tdm_load_1d(unsigned lds_addr, const void* gptr, unsigned n8)
{
    unsigned long long ga = (unsigned long long)(uintptr_t)gptr;
    u32x4 g0;
    g0.x = 1u;                                            // count=1, user desc
    g0.y = lds_addr;                                      // LDS byte address
    g0.z = (unsigned)(ga & 0xffffffffu);                  // global_addr[31:0]
    g0.w = (unsigned)((ga >> 32) & 0x01ffffffu)           // global_addr[56:32]
         | (2u << 30);                                    // type=2 ("image")
    i32x8 g1;
    g1[0] = (int)(3u << 16);                              // data_size=8B, mask=0
    g1[1] = (int)((n8 & 0xffffu) << 16);                  // tensor_dim0[15:0]
    g1[2] = (int)((n8 >> 16) & 0xffffu);                  // tensor_dim0[31:16]
    g1[3] = (int)(n8 << 16);                              // tile_dim0 = n8
    g1[4] = 0;                                            // tile_dim1=0 (1-D)
    g1[5] = (int)n8;                                      // dim0 stride (unused 1-D)
    g1[6] = 0; g1[7] = 0;
    i32x4 gz4 = {0, 0, 0, 0};
    i32x8 gz8 = {0, 0, 0, 0, 0, 0, 0, 0};
    __builtin_amdgcn_tensor_load_to_lds(g0, g1, gz4, gz4, gz8, 0 /*cpol*/);
}

// ---------------------------------------------------------------------------
// Kernel 0: repack node-net weights to f16 [outCh][K], K = tap*CPAD + c
// ---------------------------------------------------------------------------
__global__ __launch_bounds__(256)
void prep_weights_kernel(const float* __restrict__ w_n1, const float* __restrict__ w_n2,
                         _Float16* __restrict__ w1pk, _Float16* __restrict__ w2pk)
{
    const int idx = blockIdx.x * 256 + threadIdx.x;           // grid = 168 -> 43008 exact
    if (idx < HID * K1P) {
        int o = idx / K1P, k = idx - o * K1P;
        int tap = k / CP1, c = k - tap * CP1;
        float v = 0.f;
        if (k < K1R && c < 36) v = w_n1[o * 324 + c * 9 + tap];   // [64][36][3][3]
        w1pk[idx] = (_Float16)v;
    } else {
        int i2 = idx - HID * K1P;                                  // < 32*576
        int o = i2 / K2, k = i2 - o * K2;
        int tap = k >> 6, c = k & 63;
        w2pk[i2] = (_Float16)w_n2[o * 576 + c * 9 + tap];          // [32][64][3][3]
    }
}

// ---------------------------------------------------------------------------
// Kernel 1: downsampling conv (kernel==stride==2), 32ch -> 4ch, fp32
// ---------------------------------------------------------------------------
__global__ __launch_bounds__(256)
void down_kernel(const float* __restrict__ x, const float* __restrict__ w,
                 const float* __restrict__ b, float* __restrict__ xd)
{
    __shared__ float sw[512];
    __shared__ float sb[4];
    const int tid = threadIdx.x;
    for (int i = tid; i < 512; i += 256) sw[i] = w[i];  // [o][c][a][b] = o*128+c*4+a*2+b
    if (tid < 4) sb[tid] = b[tid];
    __syncthreads();
    const int gid = blockIdx.x * 256 + tid;             // N*32*32 threads
    const int n = gid >> 10, rem = gid & 1023;
    const int p = rem >> 5, q = rem & 31;
    const float* xb = x + (size_t)n * 131072 + (p * 2) * 64 + q * 2;
    float a0 = sb[0], a1 = sb[1], a2 = sb[2], a3 = sb[3];
    for (int ci = 0; ci < 32; ++ci) {
        const float* xc = xb + ci * 4096;
        float v00 = xc[0], v01 = xc[1], v10 = xc[64], v11 = xc[65];
        const float* wc = sw + ci * 4;
        a0 += v00 * wc[0]   + v01 * wc[1]   + v10 * wc[2]   + v11 * wc[3];
        a1 += v00 * wc[128] + v01 * wc[129] + v10 * wc[130] + v11 * wc[131];
        a2 += v00 * wc[256] + v01 * wc[257] + v10 * wc[258] + v11 * wc[259];
        a3 += v00 * wc[384] + v01 * wc[385] + v10 * wc[386] + v11 * wc[387];
    }
    float* op = xd + (size_t)n * 4096 + p * 32 + q;
    op[0] = a0; op[1024] = a1; op[2048] = a2; op[3072] = a3;
}

// ---------------------------------------------------------------------------
// Kernel 2: edge network. One block per edge: gather concat tile -> LDS,
// conv1(8->4)+ReLU -> LDS, conv2(4->4)+ReLU -> d_ws.  Memory-bound VALU.
// ---------------------------------------------------------------------------
__global__ __launch_bounds__(256)
void edge_kernel(const float* __restrict__ xd, const int* __restrict__ ei,
                 const float* __restrict__ w1, const float* __restrict__ b1,
                 const float* __restrict__ w2, const float* __restrict__ b2,
                 float* __restrict__ e2out)
{
    __shared__ float s_in[8 * 1024];
    __shared__ float s_mid[4 * 1024];
    __shared__ float s_w1[288], s_w2[144], s_b1[4], s_b2[4];
    const int tid = threadIdx.x, e = blockIdx.x;
    const int r = ei[e], c = ei[E_EDGE + e];
    for (int i = tid; i < 8192; i += 256) {
        int ch = i >> 10, pix = i & 1023;
        int node = (ch < 4) ? r : c;
        s_in[i] = xd[(size_t)node * 4096 + (ch & 3) * 1024 + pix];
    }
    for (int i = tid; i < 288; i += 256) s_w1[i] = w1[i];
    if (tid < 144) s_w2[tid] = w2[tid];
    if (tid < 4) { s_b1[tid] = b1[tid]; s_b2[tid] = b2[tid]; }
    __syncthreads();
    // conv1 3x3 SAME, 8->4, ReLU
    for (int t = 0; t < 4; ++t) {
        int pix = tid + (t << 8);
        int py = pix >> 5, qx = pix & 31;
        float a0 = s_b1[0], a1 = s_b1[1], a2 = s_b1[2], a3 = s_b1[3];
        for (int ci = 0; ci < 8; ++ci) {
            const float* ip = s_in + ci * 1024;
            for (int dy = -1; dy <= 1; ++dy) {
                int yy = py + dy; if ((unsigned)yy >= 32u) continue;
                for (int dx = -1; dx <= 1; ++dx) {
                    int xx = qx + dx; if ((unsigned)xx >= 32u) continue;
                    float v = ip[yy * 32 + xx];
                    int wi = ci * 9 + (dy + 1) * 3 + (dx + 1);
                    a0 += v * s_w1[wi];
                    a1 += v * s_w1[wi + 72];
                    a2 += v * s_w1[wi + 144];
                    a3 += v * s_w1[wi + 216];
                }
            }
        }
        s_mid[pix]        = a0 > 0.f ? a0 : 0.f;
        s_mid[pix + 1024] = a1 > 0.f ? a1 : 0.f;
        s_mid[pix + 2048] = a2 > 0.f ? a2 : 0.f;
        s_mid[pix + 3072] = a3 > 0.f ? a3 : 0.f;
    }
    __syncthreads();
    // conv2 3x3 SAME, 4->4, ReLU -> global
    for (int t = 0; t < 4; ++t) {
        int pix = tid + (t << 8);
        int py = pix >> 5, qx = pix & 31;
        float a0 = s_b2[0], a1 = s_b2[1], a2 = s_b2[2], a3 = s_b2[3];
        for (int ci = 0; ci < 4; ++ci) {
            const float* ip = s_mid + ci * 1024;
            for (int dy = -1; dy <= 1; ++dy) {
                int yy = py + dy; if ((unsigned)yy >= 32u) continue;
                for (int dx = -1; dx <= 1; ++dx) {
                    int xx = qx + dx; if ((unsigned)xx >= 32u) continue;
                    float v = ip[yy * 32 + xx];
                    int wi = ci * 9 + (dy + 1) * 3 + (dx + 1);
                    a0 += v * s_w2[wi];
                    a1 += v * s_w2[wi + 36];
                    a2 += v * s_w2[wi + 72];
                    a3 += v * s_w2[wi + 108];
                }
            }
        }
        float* op = e2out + (size_t)e * 4096 + pix;
        op[0]    = a0 > 0.f ? a0 : 0.f;
        op[1024] = a1 > 0.f ? a1 : 0.f;
        op[2048] = a2 > 0.f ? a2 : 0.f;
        op[3072] = a3 > 0.f ? a3 : 0.f;
    }
}

// ---------------------------------------------------------------------------
// Kernel 3: deterministic segment_sum over edges (fixed edge order, no atomics)
// ---------------------------------------------------------------------------
__global__ __launch_bounds__(256)
void agg_kernel(const float* __restrict__ e2, const int* __restrict__ row,
                float* __restrict__ agg)
{
    __shared__ int s_row[E_EDGE];
    const int tid = threadIdx.x, n = blockIdx.x;
    for (int i = tid; i < E_EDGE; i += 256) s_row[i] = row[i];
    __syncthreads();
    float acc[16];
#pragma unroll
    for (int t = 0; t < 16; ++t) acc[t] = 0.f;
    for (int e = 0; e < E_EDGE; ++e) {
        if (s_row[e] == n) {
            const float* src = e2 + (size_t)e * 4096 + tid;
#pragma unroll
            for (int t = 0; t < 16; ++t) acc[t] += src[t * 256];
        }
    }
    float* dst = agg + (size_t)n * 4096 + tid;
#pragma unroll
    for (int t = 0; t < 16; ++t) dst[t * 256] = acc[t];
}

// ---------------------------------------------------------------------------
// Kernel 4: transposed-conv upsampling (kernel==stride==2), 4->4 channels
// ---------------------------------------------------------------------------
__global__ __launch_bounds__(256)
void up_kernel(const float* __restrict__ agg, const float* __restrict__ w,
               const float* __restrict__ b, float* __restrict__ up)
{
    __shared__ float sw[64];   // [c][o][a][b] = c*16+o*4+a*2+b
    __shared__ float sb[4];
    const int tid = threadIdx.x;
    if (tid < 64) sw[tid] = w[tid];
    if (tid < 4)  sb[tid] = b[tid];
    __syncthreads();
    const int gid = blockIdx.x * 256 + tid;
    const int n = gid >> 10, rem = gid & 1023;
    const int i = rem >> 5, j = rem & 31;
    const float* ap = agg + (size_t)n * 4096 + rem;
    const float c0 = ap[0], c1 = ap[1024], c2 = ap[2048], c3 = ap[3072];
    float* ob = up + (size_t)n * 16384;
#pragma unroll
    for (int o = 0; o < 4; ++o)
#pragma unroll
        for (int a = 0; a < 2; ++a)
#pragma unroll
            for (int bb = 0; bb < 2; ++bb) {
                int wi = o * 4 + a * 2 + bb;
                float v = sb[o] + c0 * sw[wi] + c1 * sw[16 + wi]
                                + c2 * sw[32 + wi] + c3 * sw[48 + wi];
                ob[o * 4096 + (i * 2 + a) * 64 + (j * 2 + bb)] = v;
            }
}

// ---------------------------------------------------------------------------
// Kernel 5: fused node network via WMMA + TDM weight staging.
// One block = one 16x16 output tile of one image (grid = 256*16).
// GEMM1: hidden[324 x 64] = im2col(x||up)[K=360->384] * W1, +bias, ReLU
// GEMM2: out   [256 x 32] = im2col(hidden)[K=576]     * W2, +bias
// ---------------------------------------------------------------------------
__global__ __launch_bounds__(256)
void node_net_kernel(const float* __restrict__ x, const float* __restrict__ up,
                     const _Float16* __restrict__ w1pk, const _Float16* __restrict__ w2pk,
                     const float* __restrict__ b1, const float* __restrict__ b2,
                     float* __restrict__ out)
{
    extern __shared__ char smem[];
    _Float16* s_in  = (_Float16*)smem;                               // [400 px][40 ch]
    _Float16* s_w   = (_Float16*)(smem + S_IN_BYTES);                // W1 then W2
    _Float16* s_hid = (_Float16*)(smem + S_IN_BYTES + S_W_BYTES);    // [336 px][64 ch]

    const int tid  = threadIdx.x;
    const int n    = blockIdx.x >> 4;
    const int tile = blockIdx.x & 15;
    const int ty0  = (tile >> 2) * TIL;
    const int tx0  = (tile & 3) * TIL;
    const int lane = tid & 31;        // wave32
    const int wav  = tid >> 5;        // 8 waves per block
    const int g    = lane >> 4;       // half-wave id (A/B K-chunk parity)
    const int lm   = lane & 15;       // M row / N column within fragment

    // --- wave0: kick off async DMA of W1 into LDS (overlaps with staging) ---
    if (wav == 0) {
        tdm_load_1d((unsigned)S_IN_BYTES, w1pk, (unsigned)(HID * K1P * 2 / 8));
    }

    // stage input region (x channels 0..31, up channels 32..35, pad 36..39=0,
    // out-of-image halo = 0 -> implements SAME zero padding) as f16
    for (int idx = tid; idx < IR * IR * CP1; idx += 256) {
        int px = idx / CP1, c = idx - px * CP1;
        int iy = px / IR, ix = px - iy * IR;
        int gy = ty0 - 2 + iy, gx = tx0 - 2 + ix;
        float v = 0.f;
        if ((unsigned)gy < 64u && (unsigned)gx < 64u) {
            if (c < 32)      v = x [(size_t)n * 131072 + c * 4096 + gy * 64 + gx];
            else if (c < 36) v = up[(size_t)n * 16384 + (c - 32) * 4096 + gy * 64 + gx];
        }
        s_in[idx] = (_Float16)v;
    }
    if (wav == 0) __builtin_amdgcn_s_wait_tensorcnt(0);
    __syncthreads();

    // ======================= GEMM1: input -> hidden =======================
    for (int p = wav; p < 21 * 4; p += 8) {           // 21 M-tiles x 4 N-tiles
        const int mt = p >> 2, nt = p & 3;
        int m = mt * 16 + lm; if (m > 323) m = 323;   // clamp pad rows (A only)
        const int hy = m / HR, hx = m - hy * HR;
        const _Float16* bcol = s_w + (nt * 16 + lm) * K1P;
        float8 acc = {};
        // 11 full K-steps: no bounds checks in the hot loop
        for (int ks = 0; ks < 11; ++ks) {
            const int kb = ks * 32;
            Frag A, B;
            {   // A dwords 0-3: K = kb + g*8 .. +7
                int k0 = kb + (g << 3);
                int tap = k0 / CP1, c0 = k0 - tap * CP1;
                int ty = tap / 3, tx = tap - ty * 3;
                A.q[0] = *(const uint4*)(s_in + ((hy + ty) * IR + hx + tx) * CP1 + c0);
            }
            {   // A dwords 4-7: K = kb + 16 + g*8 .. +7
                int k1 = kb + 16 + (g << 3);
                int tap = k1 / CP1, c0 = k1 - tap * CP1;
                int ty = tap / 3, tx = tap - ty * 3;
                A.q[1] = *(const uint4*)(s_in + ((hy + ty) * IR + hx + tx) * CP1 + c0);
            }
            {   // B: column N = nt*16+lm, K = kb + g*8 .. +7 and kb + 16 + g*8 .. +7
                const _Float16* bp = bcol + kb + (g << 3);
                B.q[0] = *(const uint4*)bp;
                B.q[1] = *(const uint4*)(bp + 16);
            }
            acc = __builtin_amdgcn_wmma_f32_16x16x32_f16(
                      false, A.h, false, B.h, (short)0, acc, false, false);
        }
        {   // tail K-step (kb=352): only g==0 low chunk carries real K (352..359)
            Frag A, B;
            if (g == 0) {
                // K=352..359 -> tap 8, channels 32..39 (up channels + zero pad)
                A.q[0] = *(const uint4*)(s_in + ((hy + 2) * IR + hx + 2) * CP1 + 32);
            } else {
                A.q[0] = make_uint4(0, 0, 0, 0);
            }
            A.q[1] = make_uint4(0, 0, 0, 0);
            const _Float16* bp = bcol + 352 + (g << 3);
            B.q[0] = *(const uint4*)bp;
            B.q[1] = *(const uint4*)(bp + 16);
            acc = __builtin_amdgcn_wmma_f32_16x16x32_f16(
                      false, A.h, false, B.h, (short)0, acc, false, false);
        }
        const float bias = b1[nt * 16 + lm];
        // incremental row/col decode (one divide per fragment, not per element)
        int mr0 = mt * 16 + (g ? 8 : 0);
        int hy2 = mr0 / HR, hx2 = mr0 - hy2 * HR;
#pragma unroll
        for (int v = 0; v < 8; ++v) {   // D layout: VGPR v -> M = v (+8 for lanes>=16)
            int mr = mr0 + v;
            int gy = ty0 - 1 + hy2, gx = tx0 - 1 + hx2;
            float hv = acc[v] + bias;
            // true zeros outside the image so conv2's SAME padding is exact
            hv = (mr < 324 && (unsigned)gy < 64u && (unsigned)gx < 64u && hv > 0.f) ? hv : 0.f;
            s_hid[mr * CP2 + nt * 16 + lm] = (_Float16)hv;
            if (++hx2 == HR) { hx2 = 0; ++hy2; }
        }
    }
    __syncthreads();
    // --- wave0: DMA W2 over W1's LDS region ---
    if (wav == 0) {
        tdm_load_1d((unsigned)S_IN_BYTES, w2pk, (unsigned)(EMB * K2 * 2 / 8));
        __builtin_amdgcn_s_wait_tensorcnt(0);
    }
    __syncthreads();

    // ======================= GEMM2: hidden -> output ======================
    for (int p = wav; p < 16 * 2; p += 8) {           // 16 M-tiles x 2 N-tiles
        const int mt = p >> 1, nt = p & 1;
        const int m  = mt * 16 + lm;
        const int oy = m >> 4, ox = m & 15;
        const _Float16* bcol = s_w + (nt * 16 + lm) * K2;
        float8 acc = {};
        for (int ks = 0; ks < K2 / 32; ++ks) {        // 18 full steps, no guards
            const int kb = ks * 32;
            Frag A, B;
            {
                int k0 = kb + (g << 3);
                int tap = k0 >> 6, c0 = k0 & 63;
                int ty = tap / 3, tx = tap - ty * 3;
                A.q[0] = *(const uint4*)(s_hid + ((oy + ty) * HR + ox + tx) * CP2 + c0);
            }
            {
                int k1 = kb + 16 + (g << 3);
                int tap = k1 >> 6, c0 = k1 & 63;
                int ty = tap / 3, tx = tap - ty * 3;
                A.q[1] = *(const uint4*)(s_hid + ((oy + ty) * HR + ox + tx) * CP2 + c0);
            }
            {
                const _Float16* bp = bcol + kb + (g << 3);
                B.q[0] = *(const uint4*)bp;
                B.q[1] = *(const uint4*)(bp + 16);
            }
            acc = __builtin_amdgcn_wmma_f32_16x16x32_f16(
                      false, A.h, false, B.h, (short)0, acc, false, false);
        }
        const float bias = b2[nt * 16 + lm];
        const int ch = nt * 16 + lm;
#pragma unroll
        for (int v = 0; v < 8; ++v) {
            int mr = mt * 16 + (g ? 8 + v : v);
            int gy = ty0 + (mr >> 4), gx = tx0 + (mr & 15);
            out[(size_t)n * 131072 + ch * 4096 + gy * 64 + gx] = acc[v] + bias;
        }
    }
}

// ---------------------------------------------------------------------------
extern "C" void kernel_launch(void* const* d_in, const int* in_sizes, int n_in,
                              void* d_out, int out_size, void* d_ws, size_t ws_size,
                              hipStream_t stream) {
    const float* x      = (const float*)d_in[0];
    const int*   eidx   = (const int*)  d_in[1];
    /* d_in[2] batch_idx unused */
    const float* w_down = (const float*)d_in[3];
    const float* b_down = (const float*)d_in[4];
    const float* w_e1   = (const float*)d_in[5];
    const float* b_e1   = (const float*)d_in[6];
    const float* w_e2   = (const float*)d_in[7];
    const float* b_e2   = (const float*)d_in[8];
    const float* w_up   = (const float*)d_in[9];
    const float* b_up   = (const float*)d_in[10];
    const float* w_n1   = (const float*)d_in[11];
    const float* b_n1   = (const float*)d_in[12];
    const float* w_n2   = (const float*)d_in[13];
    const float* b_n2   = (const float*)d_in[14];
    float* out = (float*)d_out;

    char* ws = (char*)d_ws;
    float*     xd   = (float*)    (ws + XD_OFF);
    float*     e2b  = (float*)    (ws + E2_OFF);
    float*     agg  = (float*)    (ws + AGG_OFF);
    float*     upb  = (float*)    (ws + UP_OFF);
    _Float16*  w1pk = (_Float16*) (ws + W1_OFF);
    _Float16*  w2pk = (_Float16*) (ws + W2_OFF);

    prep_weights_kernel<<<(HID * K1P + EMB * K2) / 256, 256, 0, stream>>>(w_n1, w_n2, w1pk, w2pk);
    down_kernel<<<(N_IMG * HS * WSM) / 256, 256, 0, stream>>>(x, w_down, b_down, xd);
    edge_kernel<<<E_EDGE, 256, 0, stream>>>(xd, eidx, w_e1, b_e1, w_e2, b_e2, e2b);
    agg_kernel<<<N_IMG, 256, 0, stream>>>(e2b, eidx, agg);
    up_kernel<<<(N_IMG * HS * WSM) / 256, 256, 0, stream>>>(agg, w_up, b_up, upb);
    node_net_kernel<<<N_IMG * 16, 256, NODE_LDS, stream>>>(x, upb, w1pk, w2pk, b_n1, b_n2, out);
}
